// LigandGNN_20761871909533
// MI455X (gfx1250) — matrix operand order repaired and checked by reading, not compiled, not run on hardware
//
#include <hip/hip_runtime.h>
#include <math.h>

// ---------------- problem constants ----------------
#define Bb   32
#define Na   512
#define NBn  10
#define Ecnt 1024
#define Hd   256
#define Kh   4
#define Dd   3

typedef __attribute__((ext_vector_type(16))) __bf16 v16bf;
typedef __attribute__((ext_vector_type(8)))  float  v8f;

union Frag { v16bf v; unsigned u[8]; };

__device__ inline unsigned short f2bf(float f) {
  unsigned u = __builtin_bit_cast(unsigned, f);
  return (unsigned short)((u + 0x7fffu + ((u >> 16) & 1u)) >> 16);
}
__device__ inline unsigned pack2(float a, float b) {
  return (unsigned)f2bf(a) | ((unsigned)f2bf(b) << 16);
}

// =====================================================================
// Tiled WMMA GEMM, 128 threads (4 waves), block tile 64x64, K-step 32.
// MODE 0: logits  = reduce_j tanh(vf@Wa_main + ba)*wtilde  -> atomicAdd
// MODE 1: WLN     = gathered [vf(anb)|bond(bnb)] @ Wu2, leaky(0.1)*mask -> atomicAdd nei
// MODE 2: [vf|nei] @ Wu1, leaky(0.1) -> main_self
// MODE 3: main_self @ Wzm1 -> sigmoid gate -> vf update (GRU)
// =====================================================================
template <int MODE>
__global__ __launch_bounds__(128)
void gemm_k(const float* __restrict__ A0, const float* __restrict__ A1,
            const int*  __restrict__ I0, const int*  __restrict__ I1,
            const float* __restrict__ W,  const float* __restrict__ bias,
            const float* __restrict__ E0, const float* __restrict__ E1,
            const float* __restrict__ msk, float* __restrict__ out)
{
  constexpr int NCOLS = (MODE == 0) ? 1024 : 256;
  constexpr int KTOT  = (MODE == 1) ? 288 : ((MODE == 2) ? 512 : 256);
  constexpr int KREAL = (MODE == 1) ? 262 : KTOT;
  constexpr int NB_N  = NCOLS / 64;
  const int mb   = blockIdx.x / NB_N;
  const int nb   = blockIdx.x % NB_N;
  const int tid  = threadIdx.x;
  const int wave = tid >> 5, lane = tid & 31;
  const int half = lane >> 4, l16 = lane & 15;

  __shared__ unsigned short sA[64 * 34];   // [row][k] bf16
  __shared__ unsigned short sB[64 * 34];   // [col][k] bf16 (K-contiguous)

  v8f acc[4];
#pragma unroll
  for (int t = 0; t < 4; ++t)
#pragma unroll
    for (int e = 0; e < 8; ++e) acc[t][e] = 0.f;

  for (int k0 = 0; k0 < KTOT; k0 += 32) {
    __syncthreads();
    // ---- fill A tile: 64 rows x 16 dword pairs ----
    for (int t = tid; t < 1024; t += 128) {
      int row = t >> 4;
      int cp  = (t & 15) << 1;
      int gr  = mb * 64 + row;
      int kk  = k0 + cp;
      float f0 = 0.f, f1 = 0.f;
      if (MODE == 0 || MODE == 3) {
        const float* p = A0 + (size_t)gr * Hd + kk;
        f0 = p[0]; f1 = p[1];
      } else if (MODE == 2) {
        const float* p = (kk < Hd) ? (A0 + (size_t)gr * Hd + kk)
                                   : (A1 + (size_t)gr * Hd + (kk - Hd));
        f0 = p[0]; f1 = p[1];
      } else { // MODE 1: gathered neighbor row
        int bn = gr / NBn;
        int b  = bn >> 9;
        if (kk < Hd) {
          int ai = I0[gr];
          const float* p = A0 + ((size_t)(b * Na + ai)) * Hd + kk;
          f0 = p[0]; f1 = p[1];
        } else if (kk < KREAL) {
          int bi = I1[gr];
          int o  = kk - Hd;
          const float* p = A1 + ((size_t)(b * Ecnt + bi)) * 6;
          f0 = p[o];
          f1 = (o + 1 < 6) ? p[o + 1] : 0.f;
        }
      }
      *(unsigned*)&sA[row * 34 + cp] = pack2(f0, f1);
    }
    // ---- fill B tile: 64 cols x 16 k-pairs (transposed: K contiguous) ----
    for (int t = tid; t < 1024; t += 128) {
      int n  = t >> 4;
      int kp = (t & 15) << 1;
      int gc = nb * 64 + n;
      int kk = k0 + kp;
      float f0, f1;
      if (MODE == 0) { // Wa_main[d][k][h][j], col c = k*H+j
        int kw = gc >> 8, j = gc & 255;
        const float* p = W + (size_t)kw * Hd * Hd + (size_t)kk * Hd + j;
        f0 = p[0]; f1 = p[Hd];
      } else {
        f0 = (kk     < KREAL) ? W[(size_t)kk * Hd + gc]       : 0.f;
        f1 = (kk + 1 < KREAL) ? W[(size_t)(kk + 1) * Hd + gc] : 0.f;
      }
      *(unsigned*)&sB[n * 34 + kp] = pack2(f0, f1);
    }
    __syncthreads();

    // ---- fragments + WMMA ----
    Frag a;
#pragma unroll
    for (int i = 0; i < 8; ++i) {
      int K = ((i < 4) ? (2 * i) : (16 + 2 * (i - 4))) + 8 * half;
      a.u[i] = *(const unsigned*)&sA[(wave * 16 + l16) * 34 + K];
    }
#pragma unroll
    for (int t = 0; t < 4; ++t) {
      Frag b;
#pragma unroll
      for (int i = 0; i < 8; ++i) {
        int K = ((i < 4) ? (2 * i) : (16 + 2 * (i - 4))) + 8 * half;
        b.u[i] = *(const unsigned*)&sB[(t * 16 + l16) * 34 + K];
      }
      acc[t] = __builtin_amdgcn_wmma_f32_16x16x32_bf16(
          false, a.v, false, b.v, (short)0, acc[t], false, false);
    }
    if (MODE != 1 && k0 + 32 < KTOT)
      __builtin_prefetch(A0 + (size_t)(mb * 64) * Hd + k0 + 32, 0, 1);
  }

  // ---------------- epilogues ----------------
  if (MODE == 0) {
    const int b = (mb * 64) >> 9;               // batch constant per block
    float rs[8];
#pragma unroll
    for (int i = 0; i < 8; ++i) rs[i] = 0.f;
#pragma unroll
    for (int t = 0; t < 4; ++t) {
      int gc = nb * 64 + t * 16 + l16;
      float wt = E0[b * (Kh * Hd) + gc];        // wtilde[b,k,j]
      float bs = bias[gc];                      // ba_main[d] flat
#pragma unroll
      for (int i = 0; i < 8; ++i) rs[i] += tanhf(acc[t][i] + bs) * wt;
    }
#pragma unroll
    for (int m = 1; m < 16; m <<= 1)
#pragma unroll
      for (int i = 0; i < 8; ++i) rs[i] += __shfl_xor(rs[i], m, 32);
    if (l16 == 0) {
      int kw = (nb * 64) >> 8;
#pragma unroll
      for (int i = 0; i < 8; ++i) {
        int grow  = mb * 64 + wave * 16 + i + 8 * half;
        int natom = grow & (Na - 1);
        atomicAdd(&out[((size_t)(b * Kh + kw)) * Na + natom], rs[i]);
      }
    }
  } else if (MODE == 1) {
#pragma unroll
    for (int t = 0; t < 4; ++t) {
      int j = nb * 64 + t * 16 + l16;
      float bs = bias[j];
#pragma unroll
      for (int i = 0; i < 8; ++i) {
        int r = mb * 64 + wave * 16 + i + 8 * half;
        float y = acc[t][i] + bs;
        y = (y > 0.f) ? y : 0.1f * y;
        y *= msk[r];                            // d_nbs_mask flat
        int bn = r / NBn;
        atomicAdd(&out[(size_t)bn * Hd + j], y);
      }
    }
  } else if (MODE == 2) {
#pragma unroll
    for (int t = 0; t < 4; ++t) {
      int j = nb * 64 + t * 16 + l16;
      float bs = bias[j];
#pragma unroll
      for (int i = 0; i < 8; ++i) {
        int r = mb * 64 + wave * 16 + i + 8 * half;
        float y = acc[t][i] + bs;
        out[(size_t)r * Hd + j] = (y > 0.f) ? y : 0.1f * y;
      }
    }
  } else { // MODE 3: GRU main update
#pragma unroll
    for (int t = 0; t < 4; ++t) {
      int j = nb * 64 + t * 16 + l16;
#pragma unroll
      for (int i = 0; i < 8; ++i) {
        int r = mb * 64 + wave * 16 + i + 8 * half;
        int b = r >> 9;
        float z  = 1.f / (1.f + __expf(-(acc[t][i] + E0[b * Hd + j])));
        float ms = A0[(size_t)r * Hd + j];
        float st = E1[b * Hd + j];
        out[(size_t)r * Hd + j] = (1.f - z) * ms + z * st;
      }
    }
  }
}

// ---------------- small kernels ----------------
__global__ __launch_bounds__(256)
void embed_k(const float* __restrict__ atom, const float* __restrict__ Wv,
             const float* __restrict__ bv, float* __restrict__ vf) {
  __shared__ float sa[82];
  int bn = blockIdx.x, j = threadIdx.x;
  if (j < 82) sa[j] = atom[(size_t)bn * 82 + j];
  __syncthreads();
  float acc = bv[j];
  for (int a = 0; a < 82; ++a) acc += sa[a] * Wv[a * Hd + j];
  vf[(size_t)bn * Hd + j] = (acc > 0.f) ? acc : 0.01f * acc;
}

__global__ __launch_bounds__(256)
void sfsum_k(const float* __restrict__ vf, const float* __restrict__ dmask,
             float* __restrict__ sf) {
  __shared__ float sm[Na];
  int b = blockIdx.x, h = threadIdx.x;
  sm[h] = dmask[b * Na + h];
  sm[h + 256] = dmask[b * Na + h + 256];
  __syncthreads();
  float acc = 0.f;
  for (int n = 0; n < Na; ++n)
    acc += vf[((size_t)b * Na + n) * Hd + h] * sm[n];
  sf[b * Hd + h] = acc;
}

__global__ __launch_bounds__(256)
void superpre_k(const float* __restrict__ sf,
                const float* __restrict__ WaS, const float* __restrict__ baS,
                const float* __restrict__ Wbmm,
                const float* __restrict__ Ws2m, const float* __restrict__ bs2m,
                const float* __restrict__ Wsup, const float* __restrict__ bsup,
                const float* __restrict__ Wzm2, const float* __restrict__ bzm2,
                const float* __restrict__ bzm1,
                float* __restrict__ wtilde, float* __restrict__ stm,
                float* __restrict__ ssf, float* __restrict__ qm) {
  __shared__ float s[Hd], st[Hd];
  int b = blockIdx.x, j = threadIdx.x;
  s[j] = sf[b * Hd + j];
  __syncthreads();
  for (int k = 0; k < Kh; ++k) {
    float a = baS[k * Hd + j];
    for (int h = 0; h < Hd; ++h) a += s[h] * WaS[((size_t)k * Hd + h) * Hd + j];
    wtilde[((size_t)b * Kh + k) * Hd + j] = tanhf(a) * Wbmm[k * Hd + j];
  }
  float a2 = bs2m[j];
  for (int h = 0; h < Hd; ++h) a2 += s[h] * Ws2m[h * Hd + j];
  float stv = tanhf(a2);
  st[j] = stv; stm[b * Hd + j] = stv;
  float a3 = bsup[j];
  for (int h = 0; h < Hd; ++h) a3 += s[h] * Wsup[h * Hd + j];
  ssf[b * Hd + j] = tanhf(a3);
  __syncthreads();
  float a4 = bzm1[j] + bzm2[j];
  for (int h = 0; h < Hd; ++h) a4 += st[h] * Wzm2[h * Hd + j];
  qm[b * Hd + j] = a4;
}

__global__ __launch_bounds__(256)
void attn_k(const float* __restrict__ logits, const float* __restrict__ bbmm,
            const float* __restrict__ dmask, const float* __restrict__ vf,
            const float* __restrict__ Wm, const float* __restrict__ bm,
            const float* __restrict__ Wm2s, const float* __restrict__ bm2s,
            float* __restrict__ mts) {
  __shared__ float at[Kh][Na];
  __shared__ float p[Kh][Hd];
  __shared__ float m2s[Kh * Hd];
  __shared__ float red[256];
  __shared__ float ssum[Kh];
  int b = blockIdx.x, t = threadIdx.x;
  for (int k = 0; k < Kh; ++k) {
    float mx = -1e30f;
    for (int n = t; n < Na; n += 256) {
      float l = logits[((size_t)b * Kh + k) * Na + n] + bbmm[k];
      at[k][n] = l;
      mx = fmaxf(mx, l);
    }
    red[t] = mx; __syncthreads();
    for (int s2 = 128; s2 > 0; s2 >>= 1) {
      if (t < s2) red[t] = fmaxf(red[t], red[t + s2]);
      __syncthreads();
    }
    mx = red[0]; __syncthreads();
    float sm = 0.f;
    for (int n = t; n < Na; n += 256) {
      float e = __expf(at[k][n] - mx) * dmask[b * Na + n];
      at[k][n] = e; sm += e;
    }
    red[t] = sm; __syncthreads();
    for (int s2 = 128; s2 > 0; s2 >>= 1) {
      if (t < s2) red[t] += red[t + s2];
      __syncthreads();
    }
    float inv = 1.f / (red[0] + 1e-6f); __syncthreads();
    float ss = 0.f;
    for (int n = t; n < Na; n += 256) { at[k][n] *= inv; ss += at[k][n]; }
    red[t] = ss; __syncthreads();
    for (int s2 = 128; s2 > 0; s2 >>= 1) {
      if (t < s2) red[t] += red[t + s2];
      __syncthreads();
    }
    if (t == 0) ssum[k] = red[0];
    __syncthreads();
  }
  // pooled features p[k][h] = sum_n attn * vf  (exact linear pull-through of Wm)
  for (int k = 0; k < Kh; ++k) {
    float acc = 0.f;
    for (int n = 0; n < Na; ++n)
      acc += at[k][n] * vf[((size_t)(b * Na) + n) * Hd + t];
    p[k][t] = acc;
  }
  __syncthreads();
  for (int c = t; c < Kh * Hd; c += 256) {
    int k = c >> 8, j = c & 255;
    float acc = ssum[k] * bm[k * Hd + j];
    for (int h = 0; h < Hd; ++h) acc += p[k][h] * Wm[((size_t)k * Hd + h) * Hd + j];
    m2s[c] = acc;
  }
  __syncthreads();
  float acc = bm2s[t];
  for (int c = 0; c < Kh * Hd; ++c) acc += m2s[c] * Wm2s[(size_t)c * Hd + t];
  mts[b * Hd + t] = tanhf(acc);
}

__global__ __launch_bounds__(256)
void supup_k(const float* __restrict__ ssf, const float* __restrict__ mts,
             const float* __restrict__ Wzs1, const float* __restrict__ bzs1,
             const float* __restrict__ Wzs2, const float* __restrict__ bzs2,
             float* __restrict__ sf) {
  __shared__ float s1[Hd], s2[Hd];
  int b = blockIdx.x, j = threadIdx.x;
  s1[j] = ssf[b * Hd + j];
  s2[j] = mts[b * Hd + j];
  __syncthreads();
  float a = bzs1[j] + bzs2[j];
  for (int h = 0; h < Hd; ++h) a += s1[h] * Wzs1[h * Hd + j] + s2[h] * Wzs2[h * Hd + j];
  float z = 1.f / (1.f + __expf(-a));
  sf[b * Hd + j] = (1.f - z) * s1[j] + z * s2[j];
}

// ---------------- host orchestration ----------------
extern "C" void kernel_launch(void* const* d_in, const int* in_sizes, int n_in,
                              void* d_out, int out_size, void* d_ws, size_t ws_size,
                              hipStream_t stream) {
  const float* atom  = (const float*)d_in[0];
  const float* bond  = (const float*)d_in[1];
  const int*   anb   = (const int*)d_in[2];
  const int*   bnb   = (const int*)d_in[3];
  const float* nbs   = (const float*)d_in[4];
  const float* dmask = (const float*)d_in[5];
  const float* Wv = (const float*)d_in[6],  *bv = (const float*)d_in[7];
  const float* Wa_main = (const float*)d_in[8],  *ba_main = (const float*)d_in[9];
  const float* Wa_sup  = (const float*)d_in[10], *ba_sup  = (const float*)d_in[11];
  const float* Wm   = (const float*)d_in[12], *bm   = (const float*)d_in[13];
  const float* Wbmm = (const float*)d_in[14], *bbmm = (const float*)d_in[15];
  const float* Wm2s = (const float*)d_in[16], *bm2s = (const float*)d_in[17];
  const float* Ws2m = (const float*)d_in[18], *bs2m = (const float*)d_in[19];
  const float* Wsup = (const float*)d_in[20], *bsup = (const float*)d_in[21];
  const float* Wzm1 = (const float*)d_in[22], *bzm1 = (const float*)d_in[23];
  const float* Wzm2 = (const float*)d_in[24], *bzm2 = (const float*)d_in[25];
  const float* Wzs1 = (const float*)d_in[26], *bzs1 = (const float*)d_in[27];
  const float* Wzs2 = (const float*)d_in[28], *bzs2 = (const float*)d_in[29];
  const float* Wu2  = (const float*)d_in[30], *bu2  = (const float*)d_in[31];
  const float* Wu1  = (const float*)d_in[32], *bu1  = (const float*)d_in[33];

  float* vf = (float*)d_out;                       // [B,N,H]
  float* sf = vf + (size_t)Bb * Na * Hd;           // [B,1,H]

  float* ws = (float*)d_ws;
  float* logits = ws; ws += (size_t)Bb * Kh * Na;
  float* wtilde = ws; ws += (size_t)Bb * Kh * Hd;
  float* stm    = ws; ws += (size_t)Bb * Hd;
  float* ssfb   = ws; ws += (size_t)Bb * Hd;
  float* qm     = ws; ws += (size_t)Bb * Hd;
  float* mts    = ws; ws += (size_t)Bb * Hd;
  float* nei    = ws; ws += (size_t)Bb * Na * Hd;
  float* msbuf  = ws; ws += (size_t)Bb * Na * Hd;

  embed_k<<<Bb * Na, 256, 0, stream>>>(atom, Wv, bv, vf);
  sfsum_k<<<Bb, 256, 0, stream>>>(vf, dmask, sf);

  for (int d = 0; d < Dd; ++d) {
    size_t oKHH = (size_t)d * Kh * Hd * Hd, oKH = (size_t)d * Kh * Hd;
    size_t oHH = (size_t)d * Hd * Hd, oH = (size_t)d * Hd, oK = (size_t)d * Kh;
    hipMemsetAsync(logits, 0, (size_t)Bb * Kh * Na * sizeof(float), stream);
    hipMemsetAsync(nei, 0, (size_t)Bb * Na * Hd * sizeof(float), stream);

    superpre_k<<<Bb, 256, 0, stream>>>(sf, Wa_sup + oKHH, ba_sup + oKH, Wbmm + oKH,
                                       Ws2m + oHH, bs2m + oH, Wsup + oHH, bsup + oH,
                                       Wzm2 + oHH, bzm2 + oH, bzm1 + oH,
                                       wtilde, stm, ssfb, qm);
    // attention logits: fused tanh + weighted column reduction
    gemm_k<0><<<256 * 16, 128, 0, stream>>>(vf, nullptr, nullptr, nullptr,
                                            Wa_main + oKHH, ba_main + oKH,
                                            wtilde, nullptr, nullptr, logits);
    attn_k<<<Bb, 256, 0, stream>>>(logits, bbmm + oK, dmask, vf,
                                   Wm + oKHH, bm + oKH,
                                   Wm2s + (size_t)d * Kh * Hd * Hd, bm2s + oH, mts);
    // WLN neighbor MLP with fused gather (K = 262 padded to 288)
    gemm_k<1><<<2560 * 4, 128, 0, stream>>>(vf, bond, anb, bnb,
                                            Wu2 + (size_t)d * (Hd + 6) * Hd, bu2 + oH,
                                            nullptr, nullptr, nbs, nei);
    // main_self = leaky([vf|nei] @ Wu1)
    gemm_k<2><<<256 * 4, 128, 0, stream>>>(vf, nei, nullptr, nullptr,
                                           Wu1 + (size_t)d * 2 * Hd * Hd, bu1 + oH,
                                           nullptr, nullptr, nullptr, msbuf);
    // GRU update of vf
    gemm_k<3><<<256 * 4, 128, 0, stream>>>(msbuf, nullptr, nullptr, nullptr,
                                           Wzm1 + oHH, nullptr,
                                           qm, stm, nullptr, vf);
    supup_k<<<Bb, 256, 0, stream>>>(ssfb, mts, Wzs1 + oHH, bzs1 + oH,
                                    Wzs2 + oHH, bzs2 + oH, sf);
  }
}
